// SSMMFastBranch_9869834846609
// MI455X (gfx1250) — matrix operand order repaired and checked by reading, not compiled, not run on hardware
//
#include <hip/hip_runtime.h>
#include <hip/hip_bf16.h>

typedef __attribute__((ext_vector_type(16))) _Float16     v16h;
typedef __attribute__((ext_vector_type(8)))  float        v8f;
typedef __attribute__((ext_vector_type(4)))  unsigned int u32x4;
typedef __attribute__((ext_vector_type(8)))  unsigned int u32x8;

#define BSZ 4
#define LSZ 1024
#define HSZ 64

// ---------------------------------------------------------------------------
// Kernel 1: build per-batch conv taps (d = 0..L-1)
//   K1[b,d] = sum_h (W_out*g*W_in)_h * A_h^d   -> written as f16 (kh) for WMMA
//   K2[b,d] = sum_h (W_out*g*b_in)_h * A_h^d   -> f32, prefix-summed later
// Also emits xh = f16(x) so kernel 3 can TDM-stage both rows straight to LDS.
// grid (L/128, B), block 128
// ---------------------------------------------------------------------------
__global__ __launch_bounds__(128)
void ssm_build_taps(const float* __restrict__ x,
                    const float* __restrict__ A_s, const float* __restrict__ g_s,
                    const float* __restrict__ W_in, const float* __restrict__ b_in,
                    const float* __restrict__ W_out,
                    _Float16* __restrict__ xh, _Float16* __restrict__ kh,
                    float* __restrict__ K2) {
    __shared__ float sc1[HSZ], sc2[HSZ], sl2a[HSZ];
    const int b   = blockIdx.y;
    const int tid = threadIdx.x;
    if (tid < HSZ) {
        float g  = g_s[b * HSZ + tid];
        float w2 = W_out[tid];
        float a  = A_s[b * HSZ + tid];
        sc1[tid] = w2 * g * W_in[tid];
        sc2[tid] = w2 * g * b_in[tid];
        // a in [0,1): log2(a) -> -inf for a==0 gives exp2(-inf)=0 for d>0
        sl2a[tid] = (a > 0.0f) ? __builtin_amdgcn_logf(a) : -__builtin_inff();
    }
    __syncthreads();
    const int   d  = blockIdx.x * blockDim.x + tid;
    const float df = (float)d;
    float k1 = 0.0f, k2 = 0.0f;
    #pragma unroll 8
    for (int h = 0; h < HSZ; ++h) {
        // avoid 0 * -inf at d==0
        float pw = (d == 0) ? 1.0f : __builtin_amdgcn_exp2f(df * sl2a[h]);
        k1 = __fmaf_rn(sc1[h], pw, k1);
        k2 = __fmaf_rn(sc2[h], pw, k2);
    }
    kh[b * LSZ + d] = (_Float16)k1;
    K2[b * LSZ + d] = k2;
    xh[b * LSZ + d] = (_Float16)x[b * LSZ + d];
}

// ---------------------------------------------------------------------------
// Kernel 2: in-place inclusive prefix sum of K2 per batch (P2[b,i] = sum_{d<=i})
// grid B, block 32 (one wave). Each lane owns a contiguous 32-element chunk.
// ---------------------------------------------------------------------------
__global__ __launch_bounds__(32)
void ssm_prefix(float* __restrict__ K2) {
    __shared__ float csum[32];
    __shared__ float cbase[32];
    const int b    = blockIdx.x;
    const int lane = threadIdx.x;
    float* p = K2 + b * LSZ;
    const int C = LSZ / 32; // 32
    float s = 0.0f;
    for (int e = 0; e < C; ++e) s += p[lane * C + e];
    csum[lane] = s;
    __syncthreads();
    if (lane == 0) {
        float r = 0.0f;
        for (int i = 0; i < 32; ++i) { cbase[i] = r; r += csum[i]; }
    }
    __syncthreads();
    float r = cbase[lane];
    for (int e = 0; e < C; ++e) { r += p[lane * C + e]; p[lane * C + e] = r; }
}

// ---------------------------------------------------------------------------
// TDM helper: issue one tensor_load_to_lds of `span` contiguous f16 elements
// (1-row tile) from global `gptr` into LDS byte address `lds_byte_addr`.
// D# built per CDNA5 ISA; tracked by TENSORcnt.
// ---------------------------------------------------------------------------
__device__ __forceinline__
void tdm_load_row_f16(const void* gptr, unsigned lds_byte_addr, unsigned span) {
    unsigned long long ga = (unsigned long long)gptr;
    u32x4 g0;
    g0[0] = 1u;                                              // count=1 (valid user D#)
    g0[1] = lds_byte_addr;                                   // lds_addr [63:32]
    g0[2] = (unsigned)(ga & 0xFFFFFFFFu);                    // global_addr[31:0]
    g0[3] = (unsigned)((ga >> 32) & 0x01FFFFFFu)             // global_addr[56:32]
          | (2u << 30);                                      // type=2 ("image")
    u32x8 g1;
    g1[0] = 1u << 16;                // workgroup_mask=0 | data_size=1 (2 bytes)
    g1[1] = (span & 0xFFFFu) << 16;  // tensor_dim0[15:0]  (bits 63:48)
    g1[2] = (span >> 16)             // tensor_dim0[31:16] (bits 79:64)
          | (1u << 16);              // tensor_dim1=1      (bits 111:80 low half)
    g1[3] = (span & 0xFFFFu) << 16;  // tile_dim0 = span   (bits 127:112)
    g1[4] = 1u;                      // tile_dim1=1, tile_dim2=0
    g1[5] = span;                    // tensor_dim0_stride[31:0]
    g1[6] = 0u;                      // stride0[47:32]=0, stride1[15:0]=0
    g1[7] = 0u;                      // stride1[47:16]=0
    asm volatile("tensor_load_to_lds %0, %1" :: "s"(g0), "s"(g1) : "memory");
}

// ---------------------------------------------------------------------------
// Kernel 3: causal block-Toeplitz conv via WMMA.
//   s[b, 16p+q] = b_out + P2[b,16p+q] + sum_l Blk[l][q,:] . x_block[p-l]
//   Blk[l][q,k] = K1[16l + q - k]   (zero for negative index via LDS apron)
// One wave per (tile t, batch b): outputs i in [256t, 256t+256).
// Staging: TDM (tensor_load_to_lds) streams pre-converted f16 x and K1 rows
// into LDS; wave zeroes the causal aprons meanwhile, then s_wait_tensorcnt.
// Two lags per v_wmma_f32_16x16x32_f16. grid (4, B), block 32.
// ---------------------------------------------------------------------------
__global__ __launch_bounds__(32)
void ssm_conv_wmma(const _Float16* __restrict__ xh, const _Float16* __restrict__ kh,
                   const float* __restrict__ P2, const float* __restrict__ b_out,
                   float* __restrict__ out) {
    constexpr int XOFF = 256; // left zero apron for shifted x blocks
    constexpr int KOFF = 16;  // zero apron for causal (negative d) taps
    __shared__ _Float16 xbuf[XOFF + LSZ];
    __shared__ _Float16 kbuf[KOFF + LSZ];

    const int t    = blockIdx.x;      // output tile: blocks p in [16t, 16t+16)
    const int b    = blockIdx.y;
    const int lane = threadIdx.x;     // 0..31, wave32
    const int p0   = 16 * t;
    const int span = 256 * (t + 1);   // x / K1 elements this tile can touch

    // Async global->LDS staging via the Tensor Data Mover (TENSORcnt).
    const unsigned xlds = (unsigned)(unsigned long long)&xbuf[XOFF];
    const unsigned klds = (unsigned)(unsigned long long)&kbuf[KOFF];
    tdm_load_row_f16(xh + b * LSZ, xlds, (unsigned)span);
    tdm_load_row_f16(kh + b * LSZ, klds, (unsigned)span);

    // Zero the causal aprons while the TDM streams.
    for (int i = lane; i < XOFF; i += 32) xbuf[i] = (_Float16)0.0f;
    if (lane < KOFF) kbuf[lane] = (_Float16)0.0f;

    __builtin_amdgcn_s_wait_tensorcnt(0);
    __syncthreads();

    v8f acc = {};
    const int q   = lane & 15;               // A-matrix row (M), per ISA layout
    const int n   = lane & 15;               // B/D column (N)
    const int hiA = (lane >= 16) ? 8  : 0;   // A: upper lanes hold K+8 halves
    const int hiB = (lane >= 16) ? 8  : 0;   // B: upper lanes hold K+8 halves

    const int npairs = 8 * (t + 1);          // lags 0 .. 16t+15, two per WMMA
    for (int ip = 0; ip < npairs; ++ip) {
        const int l = 2 * ip;
        v16h av, bv;
        const int abase = KOFF + 16 * l + q;
        #pragma unroll
        for (int e = 0; e < 16; ++e) {
            // A element e -> K index per 16-bit A 16x32 layout
            int k = e + ((e < 8) ? 0 : 8) + hiA;
            // k<16: lag l, d = 16l+q-k ; k>=16: lag l+1, d = 16l+32+q-k
            av[e] = kbuf[abase - k + ((k < 16) ? 0 : 32)];
        }
        const int bbase = XOFF + 16 * (p0 + n) - 16 * l;
        #pragma unroll
        for (int e = 0; e < 16; ++e) {
            int k = e + ((e < 8) ? 0 : 8) + hiB;
            // k<16: x_block[p0+n-l][k] ; k>=16: x_block[p0+n-l-1][k-16]
            bv[e] = xbuf[bbase + k - ((k < 16) ? 0 : 32)];
        }
        acc = __builtin_amdgcn_wmma_f32_16x16x32_f16(
            /*neg_a=*/false, av, /*neg_b=*/false, bv,
            /*c_mod=*/(short)0, acc, /*reuse_a=*/false, /*reuse_b=*/false);
    }

    const float bo = b_out[0];
    #pragma unroll
    for (int e = 0; e < 8; ++e) {
        int m = e + ((lane < 16) ? 0 : 8);   // D f32 16x16 layout
        int i = 16 * (p0 + n) + m;
        out[b * LSZ + i] = acc[e] + P2[b * LSZ + i] + bo;
    }
}

// ---------------------------------------------------------------------------
extern "C" void kernel_launch(void* const* d_in, const int* in_sizes, int n_in,
                              void* d_out, int out_size, void* d_ws, size_t ws_size,
                              hipStream_t stream) {
    const float* x_f   = (const float*)d_in[0]; // (B,L,1)
    const float* A_s   = (const float*)d_in[1]; // (B,H)
    const float* g_s   = (const float*)d_in[2]; // (B,H)
    const float* W_in  = (const float*)d_in[3]; // (H,1)
    const float* b_in  = (const float*)d_in[4]; // (H,)
    const float* W_out = (const float*)d_in[5]; // (1,H)
    const float* b_out = (const float*)d_in[6]; // (1,)
    float* out = (float*)d_out;                 // (B,L,1)

    // Workspace: [0,16K) K2 f32 (prefixed in place), [16K,24K) xh f16, [24K,32K) kh f16
    float*    K2 = (float*)d_ws;
    _Float16* xh = (_Float16*)((char*)d_ws + BSZ * LSZ * sizeof(float));
    _Float16* kh = xh + BSZ * LSZ;

    ssm_build_taps<<<dim3(LSZ / 128, BSZ), 128, 0, stream>>>(
        x_f, A_s, g_s, W_in, b_in, W_out, xh, kh, K2);
    ssm_prefix<<<BSZ, 32, 0, stream>>>(K2);
    ssm_conv_wmma<<<dim3(4, BSZ), 32, 0, stream>>>(xh, kh, K2, b_out, out);
}
